// SinkhornDistance_1838246003014
// MI455X (gfx1250) — compile-verified, matrix operand order checked by reading, not run on hardware
//
#include <hip/hip_runtime.h>

// Sinkhorn distance on MI455X (gfx1250, wave32, WMMA).
// B=4, N=4096, d=64. K matrix recomputed on the fly each matvec via
// v_wmma_f32_16x16x32_f16 (compute-roofline beats streaming 268MB of stored K).

static constexpr int kB = 4;
static constexpr int kN = 4096;
static constexpr int kD = 64;
static constexpr float kEPS  = 1e-3f;
static constexpr float kSTAB = 1e-8f;
static constexpr int kITER = 100;

typedef __attribute__((ext_vector_type(16))) _Float16 v16h;
typedef __attribute__((ext_vector_type(8)))  _Float16 v8h;
typedef __attribute__((ext_vector_type(8)))  float    v8f;

__device__ __forceinline__ v16h cat8(v8h a, v8h b) {
    return __builtin_shufflevector(a, b, 0,1,2,3,4,5,6,7,8,9,10,11,12,13,14,15);
}

// Load a 16x32 f16 A/B operand tile in the CDNA5 WMMA VGPR layout.
// rows = base row of the 16-row tile in a [*, kD] row-major f16 matrix,
// kt = which 32-wide K slice (0 or 1). Lane layout (ISA 7.12.2, 16-bit A):
//   lanes 0-15  : M=lane,    K = {0..7, 16..23} (+32*kt)
//   lanes 16-31 : M=lane-16, K = {8..15, 24..31} (+32*kt)
__device__ __forceinline__ v16h load_tile(const _Float16* __restrict__ mat,
                                          size_t rowBase, int lo, int hi, int kt) {
    const _Float16* rp = mat + (rowBase + lo) * kD + kt * 32;
    v8h lo8 = *(const v8h*)(rp + hi * 8);
    v8h hi8 = *(const v8h*)(rp + 16 + hi * 8);
    return cat8(lo8, hi8);
}

// ---------------------------------------------------------------------------
// init: u = v = 1/N, out = 0
__global__ void init_uv(float* __restrict__ u, float* __restrict__ v,
                        float* __restrict__ out) {
    int i = blockIdx.x * blockDim.x + threadIdx.x;
    if (i < kB * kN) { u[i] = 1.0f / kN; v[i] = 1.0f / kN; }
    if (i == 0) out[0] = 0.0f;
}

// ---------------------------------------------------------------------------
// Row softmax over d=64; emit f16 probabilities and ||p||^2 per row.
// One wave per row; lane handles elements 2*lane, 2*lane+1 (packed b32 store).
__global__ void softmax_rows(const float* __restrict__ in,
                             _Float16* __restrict__ outh,
                             float* __restrict__ norm2) {
    int row  = blockIdx.x * (blockDim.x >> 5) + (threadIdx.x >> 5);
    int lane = threadIdx.x & 31;
    if (row >= kB * kN) return;
    const float* p = in + (size_t)row * kD;
    float a0 = p[2 * lane], a1 = p[2 * lane + 1];
    float m = fmaxf(a0, a1);
    #pragma unroll
    for (int off = 16; off >= 1; off >>= 1) m = fmaxf(m, __shfl_xor(m, off, 32));
    float e0 = __expf(a0 - m), e1 = __expf(a1 - m);
    float s = e0 + e1;
    #pragma unroll
    for (int off = 16; off >= 1; off >>= 1) s += __shfl_xor(s, off, 32);
    float inv = 1.0f / s;
    float p0 = e0 * inv, p1 = e1 * inv;
    float q = p0 * p0 + p1 * p1;
    #pragma unroll
    for (int off = 16; off >= 1; off >>= 1) q += __shfl_xor(q, off, 32);
    union { _Float16 h[2]; unsigned u32; } pk;
    pk.h[0] = (_Float16)p0; pk.h[1] = (_Float16)p1;
    ((unsigned*)(outh + (size_t)row * kD))[lane] = pk.u32;
    if (lane == 0) norm2[row] = q;
}

// ---------------------------------------------------------------------------
// Fused matvec: out_i = 1 / (sum_j exp(-cost_ij/eps) * vec_j + STAB)
// cost_ij = max(rn_i + cn_j - 2 * Ah_i . Bh_j, 0), recomputed per tile with WMMA.
// The transposed matvec is this kernel with (Ah,rn)<->(Bh,cn) swapped.
// grid = (N/16, B), block = 128 (4 waves share the 16-row tile over j-tiles).
static constexpr int kMVWaves = 4;

__global__ void sinkhorn_matvec(const _Float16* __restrict__ Ah,
                                const _Float16* __restrict__ Bh,
                                const float* __restrict__ rn,
                                const float* __restrict__ cn,
                                const float* __restrict__ vec,
                                float* __restrict__ outv) {
    const int b    = blockIdx.y;
    const int wave = threadIdx.x >> 5;
    const int lane = threadIdx.x & 31;
    const int lo   = lane & 15;
    const int hi   = lane >> 4;
    const int rowBase = blockIdx.x * 16;
    const size_t batchRow = (size_t)b * kN;

    __shared__ float ssum[16];
    if (threadIdx.x < 16) ssum[threadIdx.x] = 0.0f;
    __syncthreads();

    // A tile (16 x 64) is invariant over the j loop: keep resident in VGPRs.
    v16h A0 = load_tile(Ah, batchRow + rowBase, lo, hi, 0);
    v16h A1 = load_tile(Ah, batchRow + rowBase, lo, hi, 1);

    float rvals[8];
    #pragma unroll
    for (int r = 0; r < 8; ++r) rvals[r] = rn[batchRow + rowBase + 8 * hi + r];

    float racc[8] = {0.f, 0.f, 0.f, 0.f, 0.f, 0.f, 0.f, 0.f};

    for (int jt = wave; jt < kN / 16; jt += kMVWaves) {
        const int colBase = jt * 16;
        v16h B0 = load_tile(Bh, batchRow + colBase, lo, hi, 0);
        v16h B1 = load_tile(Bh, batchRow + colBase, lo, hi, 1);
        if (jt + kMVWaves < kN / 16)  // global_prefetch_b8 for next B tile
            __builtin_prefetch(Bh + (batchRow + (jt + kMVWaves) * 16 + lo) * kD, 0, 1);

        v8f c = {};
        c = __builtin_amdgcn_wmma_f32_16x16x32_f16(false, A0, false, B0,
                                                   (short)0, c, false, false);
        c = __builtin_amdgcn_wmma_f32_16x16x32_f16(false, A1, false, B1,
                                                   (short)0, c, false, false);
        float cnj = cn[batchRow + colBase + lo];
        float vj  = vec[batchRow + colBase + lo];
        #pragma unroll
        for (int r = 0; r < 8; ++r) {
            float cost = fmaxf(rvals[r] + cnj - 2.0f * c[r], 0.0f);
            float Kv   = __expf(cost * (-1.0f / kEPS));
            racc[r] = fmaf(Kv, vj, racc[r]);
        }
    }

    // Sum over the 16 N-lanes of each half (xor <= 8 stays within half).
    #pragma unroll
    for (int r = 0; r < 8; ++r) {
        float x = racc[r];
        #pragma unroll
        for (int off = 8; off >= 1; off >>= 1) x += __shfl_xor(x, off, 32);
        racc[r] = x;
    }
    if (lo == 0) {
        #pragma unroll
        for (int r = 0; r < 8; ++r) atomicAdd(&ssum[8 * hi + r], racc[r]);
    }
    __syncthreads();
    if (threadIdx.x < 16)
        outv[batchRow + rowBase + threadIdx.x] = 1.0f / (ssum[threadIdx.x] + kSTAB);
}

// ---------------------------------------------------------------------------
// Final: sum_ij u_i * K_ij * v_j * cost_ij, cost/K recomputed per tile.
// grid = (N/16/4, N/16, B), block = 128; each wave owns one 16x16 tile.
__global__ void sinkhorn_final(const _Float16* __restrict__ xh,
                               const _Float16* __restrict__ yh,
                               const float* __restrict__ x2,
                               const float* __restrict__ y2,
                               const float* __restrict__ u,
                               const float* __restrict__ v,
                               float* __restrict__ out) {
    const int b    = blockIdx.z;
    const int wave = threadIdx.x >> 5;
    const int lane = threadIdx.x & 31;
    const int lo   = lane & 15;
    const int hi   = lane >> 4;
    const int rowBase = blockIdx.y * 16;
    const int colBase = (blockIdx.x * 4 + wave) * 16;
    const size_t batchRow = (size_t)b * kN;

    __shared__ float bsum;
    if (threadIdx.x == 0) bsum = 0.0f;
    __syncthreads();

    v16h A0 = load_tile(xh, batchRow + rowBase, lo, hi, 0);
    v16h A1 = load_tile(xh, batchRow + rowBase, lo, hi, 1);
    v16h B0 = load_tile(yh, batchRow + colBase, lo, hi, 0);
    v16h B1 = load_tile(yh, batchRow + colBase, lo, hi, 1);

    v8f c = {};
    c = __builtin_amdgcn_wmma_f32_16x16x32_f16(false, A0, false, B0,
                                               (short)0, c, false, false);
    c = __builtin_amdgcn_wmma_f32_16x16x32_f16(false, A1, false, B1,
                                               (short)0, c, false, false);

    float cnj = y2[batchRow + colBase + lo];
    float vj  = v[batchRow + colBase + lo];
    float t = 0.0f;
    #pragma unroll
    for (int r = 0; r < 8; ++r) {
        int m = 8 * hi + r;
        float cost = fmaxf(x2[batchRow + rowBase + m] + cnj - 2.0f * c[r], 0.0f);
        float Kv = __expf(cost * (-1.0f / kEPS));
        t = fmaf(u[batchRow + rowBase + m] * Kv * vj, cost, t);
    }
    #pragma unroll
    for (int off = 16; off >= 1; off >>= 1) t += __shfl_xor(t, off, 32);
    if (lane == 0) atomicAdd(&bsum, t);
    __syncthreads();
    if (threadIdx.x == 0) atomicAdd(out, bsum);
}

// ---------------------------------------------------------------------------
extern "C" void kernel_launch(void* const* d_in, const int* in_sizes, int n_in,
                              void* d_out, int out_size, void* d_ws, size_t ws_size,
                              hipStream_t stream) {
    const float* x = (const float*)d_in[0];
    const float* y = (const float*)d_in[1];
    float* out = (float*)d_out;

    // Workspace layout (~4.5 MB total)
    char* ws = (char*)d_ws;
    _Float16* xh = (_Float16*)ws;                                  // 2 MB
    _Float16* yh = (_Float16*)(ws + (size_t)kB * kN * kD * 2);     // 2 MB
    float* x2 = (float*)(ws + (size_t)2 * kB * kN * kD * 2);
    float* y2 = x2 + kB * kN;
    float* u  = y2 + kB * kN;
    float* v  = u  + kB * kN;

    init_uv<<<(kB * kN + 255) / 256, 256, 0, stream>>>(u, v, out);
    softmax_rows<<<kB * kN / 4, 128, 0, stream>>>(x, xh, x2);
    softmax_rows<<<kB * kN / 4, 128, 0, stream>>>(y, yh, y2);

    dim3 mg(kN / 16, kB);
    for (int it = 0; it < kITER; ++it) {
        // u_i = 1/(sum_j K_ij v_j + stab)
        sinkhorn_matvec<<<mg, 128, 0, stream>>>(xh, yh, x2, y2, v, u);
        // v_j = 1/(sum_i K_ij u_i + stab)  ==  same kernel with roles swapped
        sinkhorn_matvec<<<mg, 128, 0, stream>>>(yh, xh, y2, x2, u, v);
    }

    dim3 fg(kN / 16 / 4, kN / 16, kB);
    sinkhorn_final<<<fg, 128, 0, stream>>>(xh, yh, x2, y2, u, v, out);
}
